// Bottleneck_43198781063849
// MI455X (gfx1250) — matrix operand order repaired and checked by reading, not compiled, not run on hardware
//
#include <hip/hip_runtime.h>

typedef __attribute__((ext_vector_type(16))) __bf16 v16bf;
typedef __attribute__((ext_vector_type(8)))  __bf16 v8bf;
typedef __attribute__((ext_vector_type(8)))  float  v8f;
typedef int v4i __attribute__((vector_size(16)));

#define DEVINL __device__ __forceinline__
#define SHUF16(a, b) \
  __builtin_shufflevector(a, b, 0, 1, 2, 3, 4, 5, 6, 7, 8, 9, 10, 11, 12, 13, 14, 15)

#if defined(__has_builtin)
#if __has_builtin(__builtin_amdgcn_global_load_async_to_lds_b128)
#define HAVE_ASYNC_LDS 1
#endif
#if __has_builtin(__builtin_amdgcn_s_wait_asynccnt)
#define HAVE_WAIT_ASYNC 1
#endif
#endif

DEVINL v8bf zero_v8bf() {
  v8bf z;
#pragma unroll
  for (int i = 0; i < 8; ++i) z[i] = (__bf16)0.0f;
  return z;
}

// 16B global -> LDS copy: async DMA path (ASYNCcnt) if available, else via regs
DEVINL void cp16_g2l(void* l, const void* g) {
#ifdef HAVE_ASYNC_LDS
  __builtin_amdgcn_global_load_async_to_lds_b128(
      (__attribute__((address_space(1))) v4i*)g,
      (__attribute__((address_space(3))) v4i*)l, 0, 0);
#else
  *(v8bf*)l = *(const v8bf*)g;
#endif
}

DEVINL void pipeline_wait() {
#ifdef HAVE_ASYNC_LDS
#ifdef HAVE_WAIT_ASYNC
  __builtin_amdgcn_s_wait_asynccnt(0);
#else
  asm volatile("s_wait_asynccnt 0" ::: "memory");
#endif
#endif
  __syncthreads();
}

// ---------------------------------------------------------------------------
// Implicit-GEMM conv with WMMA bf16 (D = W * im2col(X)), fused BN(+ReLU)
// A = weights bf16 [tap][MT][KG]; B = activations bf16 NHWC [B*H*W][CT],
// group g uses channels [g*KG, (g+1)*KG).
// Block: 256 threads = 8 waves. Block tile 64(M) x 64(N), double-buffered
// LDS, K-step 64 (two WMMA K-slices per stage). Wave (wm 0..3, wn 0..1)
// computes 16M x 32N via two f32 accumulators.
// ---------------------------------------------------------------------------
template <int KS, int HH, int WW, int MT, int KG, int CT, int MG,
          bool RELU, bool MASK, bool FINAL, bool OUTB>
__global__ __launch_bounds__(256) void conv_wmma(
    const __bf16* __restrict__ Wb, const __bf16* __restrict__ Bin,
    const float* __restrict__ sb,      // [scale[MT], bias[MT]]
    const float* __restrict__ maskp,   // [B][2][7][7] binary, or null
    const float* __restrict__ resid,   // x, NCHW fp32 (FINAL)
    const float* __restrict__ xb3,     // 14x14 branch, NCHW fp32 (FINAL)
    __bf16* __restrict__ outH,         // NHWC bf16 out (OUTB)
    float* __restrict__ outF) {        // NCHW fp32 out (!OUTB)
  constexpr int HW = HH * WW;
  constexpr int KSTEPS = KG / 64;
  constexpr int NS = KS * KS * KSTEPS;

  const int tid = threadIdx.x;
  const int lane = tid & 31, wave = tid >> 5;
  const int wm = wave >> 1, wn = wave & 1;
  const int lrow = lane & 15, lhi = lane >> 4;
  const int n_base = blockIdx.x * 64;
  const int oc_base = blockIdx.y * 64;
  const int g = oc_base / MG;

  __shared__ __align__(16) __bf16 Asm[2][64][72];
  __shared__ __align__(16) __bf16 Bsm[2][64][72];

  v8f acc0 = {};
  v8f acc1 = {};

  // staging: each thread moves two 16B chunks per tile (rows r0, r0+32)
  const int r0 = tid >> 3;
  const int colb = (tid & 7) * 8;
  int bb[2], by[2], bx[2];
#pragma unroll
  for (int i = 0; i < 2; ++i) {
    const int p = n_base + r0 + 32 * i;
    bb[i] = p / HW;
    const int rem = p - bb[i] * HW;
    by[i] = rem / WW;
    bx[i] = rem - by[i] * WW;
  }

  auto stage = [&](int s, int buf) {
    const int t = s / KSTEPS;
    const int kk = (s - t * KSTEPS) * 64;
    const int dy = (KS == 3) ? (t / 3 - 1) : 0;
    const int dx = (KS == 3) ? (t % 3 - 1) : 0;
#pragma unroll
    for (int i = 0; i < 2; ++i) {
      const int row = r0 + 32 * i;
      cp16_g2l(&Asm[buf][row][colb],
               Wb + (size_t)(t * MT + oc_base + row) * KG + kk + colb);
      const int yy = by[i] + dy, xx = bx[i] + dx;
      const bool valid =
          (KS == 1) || (yy >= 0 && yy < HH && xx >= 0 && xx < WW);
      if (valid)
        cp16_g2l(&Bsm[buf][row][colb],
                 Bin + (size_t)(bb[i] * HW + yy * WW + xx) * CT + g * KG + kk + colb);
      else
        *(v8bf*)&Bsm[buf][row][colb] = zero_v8bf();
    }
  };

  auto compute = [&](int buf) {
#pragma unroll
    for (int kq = 0; kq < 64; kq += 32) {
      v8bf alo = *(const v8bf*)&Asm[buf][wm * 16 + lrow][kq + lhi * 16];
      v8bf ahi = *(const v8bf*)&Asm[buf][wm * 16 + lrow][kq + lhi * 16 + 8];
      v16bf af = SHUF16(alo, ahi);
      v8bf b0l = *(const v8bf*)&Bsm[buf][wn * 32 + lrow][kq + lhi * 16];
      v8bf b0h = *(const v8bf*)&Bsm[buf][wn * 32 + lrow][kq + lhi * 16 + 8];
      v16bf bf0 = SHUF16(b0l, b0h);
      v8bf b1l = *(const v8bf*)&Bsm[buf][wn * 32 + 16 + lrow][kq + lhi * 16];
      v8bf b1h = *(const v8bf*)&Bsm[buf][wn * 32 + 16 + lrow][kq + lhi * 16 + 8];
      v16bf bf1 = SHUF16(b1l, b1h);
      acc0 = __builtin_amdgcn_wmma_f32_16x16x32_bf16(false, af, false, bf0,
                                                     (short)0, acc0, false, false);
      acc1 = __builtin_amdgcn_wmma_f32_16x16x32_bf16(false, af, false, bf1,
                                                     (short)0, acc1, false, false);
    }
  };

  stage(0, 0);
  pipeline_wait();
#pragma unroll 1
  for (int s = 0; s < NS; ++s) {
    const int cur = s & 1;
    if (s + 1 < NS) stage(s + 1, cur ^ 1);
    compute(cur);
    pipeline_wait();
  }

  // epilogue: lane holds N = lrow col of subtile, M rows oc0..oc0+7
  const int oc0 = oc_base + wm * 16 + lhi * 8;
#pragma unroll
  for (int sub = 0; sub < 2; ++sub) {
    v8f acc = sub ? acc1 : acc0;
    const int p = n_base + wn * 32 + sub * 16 + lrow;
    const int b_ = p / HW;
    const int rem = p - b_ * HW;
    const int py = rem / WW, px = rem - py * WW;
    float mval = 1.0f;
    if (MASK) mval = maskp[(b_ * 2 + g) * 49 + (py >> 2) * 7 + (px >> 2)];
    if (OUTB) {
      v8bf ov;
#pragma unroll
      for (int r = 0; r < 8; ++r) {
        const int oc = oc0 + r;
        float v = acc[r] * sb[oc] + sb[MT + oc];
        if (RELU) v = fmaxf(v, 0.0f);
        if (MASK) v *= mval;
        ov[r] = (__bf16)v;
      }
      *(v8bf*)&outH[(size_t)p * MT + oc0] = ov;
    } else {
#pragma unroll
      for (int r = 0; r < 8; ++r) {
        const int oc = oc0 + r;
        float v = acc[r] * sb[oc] + sb[MT + oc];
        if (FINAL) {
          v += resid[(size_t)(b_ * MT + oc) * HW + rem];
          v += xb3[(size_t)(b_ * MT + oc) * 196 + (py >> 1) * 14 + (px >> 1)];
          v = fmaxf(v, 0.0f);
        }
        outF[(size_t)(b_ * MT + oc) * HW + rem] = v;
      }
    }
  }
}

// ---------------------------------------------------------------------------
// Prep / small kernels
// ---------------------------------------------------------------------------

// Fold BN stack [gamma,beta,mean,var][C] -> [scale[C], bias[C]]
__global__ void bn_prep(const float* __restrict__ src, float* __restrict__ dst,
                        int Cc) {
  int c = blockIdx.x * blockDim.x + threadIdx.x;
  if (c >= Cc) return;
  float gma = src[c], bta = src[Cc + c], m = src[2 * Cc + c], v = src[3 * Cc + c];
  float s = gma * rsqrtf(v + 1e-5f);
  dst[c] = s;
  dst[Cc + c] = bta - m * s;
}

// OIHW fp32 -> [tap][oc][ic] bf16
__global__ void reorder_w(const float* __restrict__ src, __bf16* __restrict__ dst,
                          int Mtot, int Kg, int KS2) {
  int idx = blockIdx.x * blockDim.x + threadIdx.x;
  int n = Mtot * Kg * KS2;
  if (idx >= n) return;
  int t = idx / (Mtot * Kg);
  int r2 = idx - t * (Mtot * Kg);
  int oc = r2 / Kg;
  int kg = r2 - oc * Kg;
  dst[idx] = (__bf16)src[(size_t)(oc * Kg + kg) * KS2 + t];
}

// x NCHW fp32 -> bf16 NHWC [B*784][1024] (LDS-transposed for coalescing)
__global__ __launch_bounds__(256) void cvt_nhwc(const float* __restrict__ x,
                                                __bf16* __restrict__ xh) {
  __shared__ float tile[32][33];
  const int tx = threadIdx.x & 31, ty = threadIdx.x >> 5;
  const int p0 = blockIdx.x * 32, c0 = blockIdx.y * 32, b = blockIdx.z;
#pragma unroll
  for (int k = 0; k < 4; ++k) {
    int p = p0 + tx;
    if (p < 784)
      tile[ty + k * 8][tx] = x[(size_t)(b * 1024 + c0 + ty + k * 8) * 784 + p];
  }
  __syncthreads();
#pragma unroll
  for (int k = 0; k < 4; ++k) {
    int pr = p0 + ty + k * 8;
    if (pr < 784)
      xh[(size_t)(b * 784 + pr) * 1024 + c0 + tx] = (__bf16)tile[tx][ty + k * 8];
  }
}

// 3x3 s2 p1 avgpool (pad-with-zero, /9) -> bf16 NHWC [B*196][1024]
__global__ void avgpool_nhwc(const float* __restrict__ x,
                             __bf16* __restrict__ pool) {
  size_t idx = (size_t)blockIdx.x * blockDim.x + threadIdx.x;
  if (idx >= (size_t)32 * 196 * 1024) return;
  int c = idx & 1023;
  int tpi = (int)(idx >> 10);
  int p = tpi % 196, b = tpi / 196;
  int oy = p / 14, ox = p % 14;
  float s = 0.0f;
#pragma unroll
  for (int dy = -1; dy <= 1; ++dy) {
    int iy = 2 * oy + dy;
    if (iy < 0 || iy >= 28) continue;
#pragma unroll
    for (int dx = -1; dx <= 1; ++dx) {
      int ix = 2 * ox + dx;
      if (ix < 0 || ix >= 28) continue;
      s += x[(size_t)(b * 1024 + c) * 784 + iy * 28 + ix];
    }
  }
  pool[idx] = (__bf16)(s * (1.0f / 9.0f));
}

// gate grouped 3x3 conv (8 out ch) + BN + ReLU, direct VALU
__global__ void gate_conv(const __bf16* __restrict__ xh,
                          const float* __restrict__ mgw,
                          const float* __restrict__ sb,
                          float* __restrict__ gte) {
  int idx = blockIdx.x * blockDim.x + threadIdx.x;
  if (idx >= 32 * 8 * 784) return;
  int p = idx % 784;
  int oc = (idx / 784) & 7;
  int b = idx / (784 * 8);
  int y = p / 28, x = p % 28, g = oc >> 2;
  float acc = 0.0f;
  for (int t = 0; t < 9; ++t) {
    int yy = y + t / 3 - 1, xx = x + t % 3 - 1;
    if (yy < 0 || yy >= 28 || xx < 0 || xx >= 28) continue;
    const __bf16* src = xh + (size_t)(b * 784 + yy * 28 + xx) * 1024 + g * 512;
    const float* w = mgw + (size_t)oc * 512 * 9 + t;
#pragma unroll 8
    for (int ic = 0; ic < 512; ++ic) acc += (float)src[ic] * w[ic * 9];
  }
  gte[(size_t)(b * 8 + oc) * 784 + p] = fmaxf(acc * sb[oc] + sb[8 + oc], 0.0f);
}

// 4x4 mean pool -> grouped 1x1 fc + bias -> argmax -> binary mask [B][2][7][7]
__global__ void gate_mask(const float* __restrict__ gte,
                          const float* __restrict__ fcw,
                          const float* __restrict__ fcb,
                          float* __restrict__ maskout) {
  int idx = blockIdx.x * blockDim.x + threadIdx.x;
  if (idx >= 32 * 2 * 49) return;
  int pp = idx % 49;
  int g = (idx / 49) & 1;
  int b = idx / 98;
  int py = pp / 7, px = pp % 7;
  float pooled[4];
#pragma unroll
  for (int ic = 0; ic < 4; ++ic) {
    const float* src = gte + (size_t)(b * 8 + g * 4 + ic) * 784;
    float s = 0.0f;
#pragma unroll
    for (int dy = 0; dy < 4; ++dy)
#pragma unroll
      for (int dx = 0; dx < 4; ++dx)
        s += src[(py * 4 + dy) * 28 + px * 4 + dx];
    pooled[ic] = s * (1.0f / 16.0f);
  }
  float l0 = fcb[g * 2 + 0], l1 = fcb[g * 2 + 1];
#pragma unroll
  for (int ic = 0; ic < 4; ++ic) {
    l0 += pooled[ic] * fcw[(g * 2 + 0) * 4 + ic];
    l1 += pooled[ic] * fcw[(g * 2 + 1) * 4 + ic];
  }
  maskout[idx] = (l1 > l0) ? 1.0f : 0.0f;
}

// ---------------------------------------------------------------------------
// Launch
// ---------------------------------------------------------------------------
extern "C" void kernel_launch(void* const* d_in, const int* in_sizes, int n_in,
                              void* d_out, int out_size, void* d_ws,
                              size_t ws_size, hipStream_t stream) {
  const float* x       = (const float*)d_in[0];
  const float* mgw     = (const float*)d_in[1];
  const float* mg_bn   = (const float*)d_in[2];
  const float* fcw     = (const float*)d_in[3];
  const float* fcb     = (const float*)d_in[4];
  const float* c1b_w   = (const float*)d_in[5];
  const float* bn1b    = (const float*)d_in[6];
  const float* c2b_w   = (const float*)d_in[7];
  const float* bn2b    = (const float*)d_in[8];
  const float* c3b_w   = (const float*)d_in[9];
  const float* bn3b    = (const float*)d_in[10];
  const float* c1r_w   = (const float*)d_in[11];
  const float* bn1r    = (const float*)d_in[12];
  const float* c2r_w   = (const float*)d_in[13];
  const float* bn2r    = (const float*)d_in[14];
  const float* c3r_w   = (const float*)d_in[15];
  const float* bn3r    = (const float*)d_in[16];

  char* ws = (char*)d_ws;
  size_t off = 0;
  auto alloc = [&](size_t bytes) {
    char* p = ws + off;
    off = (off + bytes + 255) & ~(size_t)255;
    return p;
  };
  __bf16* xh     = (__bf16*)alloc((size_t)32 * 784 * 1024 * 2);
  __bf16* pool_h = (__bf16*)alloc((size_t)32 * 196 * 1024 * 2);
  __bf16* xb1_h  = (__bf16*)alloc((size_t)32 * 196 * 256 * 2);
  __bf16* xb2_h  = (__bf16*)alloc((size_t)32 * 196 * 256 * 2);
  float*  xb3    = (float*)alloc((size_t)32 * 196 * 1024 * 4);
  __bf16* xr1_h  = (__bf16*)alloc((size_t)32 * 784 * 256 * 2);
  __bf16* xr2_h  = (__bf16*)alloc((size_t)32 * 784 * 256 * 2);
  float*  gte    = (float*)alloc((size_t)32 * 8 * 784 * 4);
  __bf16* wb1b   = (__bf16*)alloc((size_t)256 * 1024 * 2);
  __bf16* wb2b   = (__bf16*)alloc((size_t)9 * 256 * 256 * 2);
  __bf16* wb3b   = (__bf16*)alloc((size_t)1024 * 256 * 2);
  __bf16* wb1r   = (__bf16*)alloc((size_t)256 * 512 * 2);
  __bf16* wb2r   = (__bf16*)alloc((size_t)9 * 256 * 128 * 2);
  __bf16* wb3r   = (__bf16*)alloc((size_t)1024 * 128 * 2);
  float*  sb1b   = (float*)alloc(2 * 256 * 4);
  float*  sb2b   = (float*)alloc(2 * 256 * 4);
  float*  sb3b   = (float*)alloc(2 * 1024 * 4);
  float*  sb1r   = (float*)alloc(2 * 256 * 4);
  float*  sb2r   = (float*)alloc(2 * 256 * 4);
  float*  sb3r   = (float*)alloc(2 * 1024 * 4);
  float*  sbmg   = (float*)alloc(2 * 8 * 4);

  float* out  = (float*)d_out;
  float* mask = out + (size_t)32 * 1024 * 28 * 28;  // tuple tail: (32,2,7,7)

  // BN folds
  bn_prep<<<1, 256, 0, stream>>>(bn1b, sb1b, 256);
  bn_prep<<<1, 256, 0, stream>>>(bn2b, sb2b, 256);
  bn_prep<<<4, 256, 0, stream>>>(bn3b, sb3b, 1024);
  bn_prep<<<1, 256, 0, stream>>>(bn1r, sb1r, 256);
  bn_prep<<<1, 256, 0, stream>>>(bn2r, sb2r, 256);
  bn_prep<<<4, 256, 0, stream>>>(bn3r, sb3r, 1024);
  bn_prep<<<1, 256, 0, stream>>>(mg_bn, sbmg, 8);

  // Weight reorder/convert
  auto rw = [&](const float* s, __bf16* d, int M, int K, int T) {
    int n = M * K * T;
    reorder_w<<<(n + 255) / 256, 256, 0, stream>>>(s, d, M, K, T);
  };
  rw(c1b_w, wb1b, 256, 1024, 1);
  rw(c2b_w, wb2b, 256, 256, 9);
  rw(c3b_w, wb3b, 1024, 256, 1);
  rw(c1r_w, wb1r, 256, 512, 1);
  rw(c2r_w, wb2r, 256, 128, 9);
  rw(c3r_w, wb3r, 1024, 128, 1);

  // Activation layout conversions
  cvt_nhwc<<<dim3(25, 32, 32), 256, 0, stream>>>(x, xh);
  {
    size_t n = (size_t)32 * 196 * 1024;
    avgpool_nhwc<<<(unsigned)((n + 255) / 256), 256, 0, stream>>>(x, pool_h);
  }

  // Gate path -> mask (written to output tail, then consumed by c1r/c2r)
  gate_conv<<<(32 * 8 * 784 + 255) / 256, 256, 0, stream>>>(xh, mgw, sbmg, gte);
  gate_mask<<<(32 * 2 * 49 + 255) / 256, 256, 0, stream>>>(gte, fcw, fcb, mask);

  // Downsampled branch: c1b (1x1) -> c2b (3x3) -> c3b (1x1, BN only, fp32 NCHW)
  conv_wmma<1, 14, 14, 256, 1024, 1024, 256, true, false, false, true>
      <<<dim3(98, 4), 256, 0, stream>>>(wb1b, pool_h, sb1b, nullptr, nullptr,
                                        nullptr, xb1_h, nullptr);
  conv_wmma<3, 14, 14, 256, 256, 256, 256, true, false, false, true>
      <<<dim3(98, 4), 256, 0, stream>>>(wb2b, xb1_h, sb2b, nullptr, nullptr,
                                        nullptr, xb2_h, nullptr);
  conv_wmma<1, 14, 14, 1024, 256, 256, 1024, false, false, false, false>
      <<<dim3(98, 16), 256, 0, stream>>>(wb3b, xb2_h, sb3b, nullptr, nullptr,
                                         nullptr, nullptr, xb3);

  // Masked branch: c1r (grouped 1x1, mask folds into epilogue) -> c2r (grouped
  // 3x3) -> c3r (grouped 1x1 + residual + upsampled xb3 + ReLU, final out)
  conv_wmma<1, 28, 28, 256, 512, 1024, 128, true, true, false, true>
      <<<dim3(392, 4), 256, 0, stream>>>(wb1r, xh, sb1r, mask, nullptr,
                                         nullptr, xr1_h, nullptr);
  conv_wmma<3, 28, 28, 256, 128, 256, 128, true, true, false, true>
      <<<dim3(392, 4), 256, 0, stream>>>(wb2r, xr1_h, sb2r, mask, nullptr,
                                         nullptr, xr2_h, nullptr);
  conv_wmma<1, 28, 28, 1024, 128, 256, 512, false, false, true, false>
      <<<dim3(392, 16), 256, 0, stream>>>(wb3r, xr2_h, sb3r, nullptr, x, xb3,
                                          nullptr, out);
}
